// BatchedMaskedModel_85933705658712
// MI455X (gfx1250) — compile-verified, matrix operand order checked by reading, not run on hardware
//
#include <hip/hip_runtime.h>
#include <cstdint>
#include <cstddef>

// ---------------------------------------------------------------------------
// MI455X (gfx1250) transformer block, bf16 WMMA path.
// ~70 GFLOP of GEMMs vs ~100MB traffic -> matrix-core bound; everything
// routed through v_wmma_f32_16x16x32_bf16 with f32 accumulation.
// Row-major tiles are staged into LDS with global_load_async_to_lds_b128
// (ASYNCcnt-tracked DMA, no VGPR round-trip), double-buffered in the GEMM.
// ---------------------------------------------------------------------------

typedef __attribute__((ext_vector_type(16))) __bf16 bf16x16;
typedef __attribute__((ext_vector_type(8)))  float  f32x8;

#define HEADS 8
#define DHEAD 64
#define SEQ   1024
#define BATCH 8
#define DMODEL 512
#define DFF   2048
#define NCLS  1000
#define MROWS (BATCH * SEQ)   // 8192
#define ATTN_SCALE 0.125f     // 1/sqrt(64)

__device__ __forceinline__ f32x8 zero8() {
    f32x8 z = {0.f, 0.f, 0.f, 0.f, 0.f, 0.f, 0.f, 0.f};
    return z;
}

__device__ __forceinline__ float gelu_tanh(float x) {
    float x3 = x * x * x;
    return 0.5f * x * (1.f + tanhf(0.7978845608028654f * (x + 0.044715f * x3)));
}

// Async DMA: 16 bytes global -> LDS per lane (VDST = LDS byte offset VGPR,
// VADDR = 64-bit global address). Tracked by ASYNCcnt. Low 32 bits of a
// flat shared pointer are the LDS byte offset on AMDGPU.
__device__ __forceinline__ void async_g2l_b128(void* ldsDst, const void* gSrc) {
    unsigned ldsOff = (unsigned)(uintptr_t)ldsDst;
    asm volatile("global_load_async_to_lds_b128 %0, %1, off"
                 :: "v"(ldsOff), "v"(gSrc)
                 : "memory");
}

__device__ __forceinline__ void wait_async0() {
#if __has_builtin(__builtin_amdgcn_s_wait_asynccnt)
    __builtin_amdgcn_s_wait_asynccnt(0);
#else
    asm volatile("s_wait_asynccnt 0x0" ::: "memory");
#endif
}

// Load a 16x32 bf16 WMMA operand fragment from an LDS tile laid out as
// tile[rc][k] (A-matrix rows, or B-matrix stored K-transposed: tile[col][k]).
// Per ISA 7.12.2 (16-bit A 16x32): lanes 0-15 hold K=0..7,16..23 of row M=lane,
// lanes 16-31 hold K=8..15,24..31 of row M=lane-16; pairs packed per VGPR.
__device__ __forceinline__ bf16x16 load_frag(const __bf16* base, int stride,
                                             int rcBase, int kBase) {
    const int lane = threadIdx.x & 31;
    const int idx  = lane & 15;
    const int half = lane >> 4;
    const __bf16* p = base + (size_t)(rcBase + idx) * stride + kBase + half * 8;
    union { bf16x16 v; unsigned u[8]; } r;
#pragma unroll
    for (int i = 0; i < 8; ++i) {
        int k = ((i & 4) << 2) + ((i & 3) << 1);   // i<4 -> 0..6 ; i>=4 -> 16..22
        r.u[i] = *(const unsigned*)(p + k);
    }
    return r.v;
}

// ---------------------------------------------------------------------------
// fp32 -> bf16 convert
// ---------------------------------------------------------------------------
__global__ __launch_bounds__(256) void cvt_bf16_kernel(const float* __restrict__ in,
                                                       __bf16* __restrict__ out, int n) {
    int i = blockIdx.x * 256 + threadIdx.x;
    if (i < n) out[i] = (__bf16)in[i];
}

// ---------------------------------------------------------------------------
// LayerNorm over D=512. Optionally applies token mask and emits f32 residual.
// One block (256 threads, 2 cols each) per row.
// ---------------------------------------------------------------------------
template <bool MASKED>
__global__ __launch_bounds__(256) void ln_kernel(const float* __restrict__ x,
                                                 const float* __restrict__ mask,
                                                 float* __restrict__ residOut,
                                                 const float* __restrict__ g,
                                                 const float* __restrict__ bta,
                                                 __bf16* __restrict__ outB) {
    __shared__ float red[256];
    __shared__ float stats[2];
    const int row = blockIdx.x;
    const int tid = threadIdx.x;
    const size_t rb = (size_t)row * DMODEL;
    float v0 = x[rb + tid];
    float v1 = x[rb + tid + 256];
    if (MASKED) {
        float mv = mask[row & (SEQ - 1)];
        v0 *= mv; v1 *= mv;
        residOut[rb + tid]       = v0;
        residOut[rb + tid + 256] = v1;
    }
    red[tid] = v0 + v1;
    __syncthreads();
#pragma unroll
    for (int off = 128; off > 0; off >>= 1) {
        if (tid < off) red[tid] += red[tid + off];
        __syncthreads();
    }
    if (tid == 0) stats[0] = red[0] * (1.f / DMODEL);
    __syncthreads();
    const float mu = stats[0];
    float d0 = v0 - mu, d1 = v1 - mu;
    red[tid] = d0 * d0 + d1 * d1;
    __syncthreads();
#pragma unroll
    for (int off = 128; off > 0; off >>= 1) {
        if (tid < off) red[tid] += red[tid + off];
        __syncthreads();
    }
    if (tid == 0) stats[1] = rsqrtf(red[0] * (1.f / DMODEL) + 1e-5f);
    __syncthreads();
    const float rstd = stats[1];
    outB[rb + tid]       = (__bf16)(d0 * rstd * g[tid]       + bta[tid]);
    outB[rb + tid + 256] = (__bf16)(d1 * rstd * g[tid + 256] + bta[tid + 256]);
}

// ---------------------------------------------------------------------------
// bf16 WMMA GEMM: C[M,N] = A[M,K] @ B[K,N] (+bias)(+GELU)(+resid)
// Block: 256 threads = 8 waves. Tile BM=128, BN=128, BK=32, double-buffered:
// next tile's async DMA runs under the current tile's WMMAs.
// Waves: 4 along M (32 rows), 2 along N (64 cols) -> 2x4 subtiles of 16x16.
// B is stored K-transposed in LDS so both operands use the same frag loader.
// ---------------------------------------------------------------------------
template <bool BIAS, bool GELU_ACT, bool RESID, bool OUT_BF16>
__global__ __launch_bounds__(256) void gemm_bf16_kernel(
        const __bf16* __restrict__ A, const __bf16* __restrict__ Bm,
        const float* __restrict__ bias, const float* __restrict__ resid,
        float* __restrict__ outF, __bf16* __restrict__ outB,
        int M, int Nn, int K) {
    __shared__ __bf16 lsA[2][128][40];   // [buf][row][k]
    __shared__ __bf16 lsB[2][128][40];   // [buf][col][k]  (K-transposed B tile)

    const int m0 = blockIdx.y * 128;
    const int n0 = blockIdx.x * 128;
    const int tid  = threadIdx.x;
    const int wave = tid >> 5;
    const int wm   = wave & 3;    // 0..3 : 32 rows each
    const int wn   = wave >> 2;   // 0..1 : 64 cols each

    f32x8 acc[2][4];
#pragma unroll
    for (int mi = 0; mi < 2; ++mi)
#pragma unroll
        for (int ni = 0; ni < 4; ++ni) acc[mi][ni] = zero8();

    auto load_tiles = [&](int k0, int buf) {
        // A tile: 128 rows x 32 cols, async DMA straight to LDS
#pragma unroll
        for (int s = tid; s < 512; s += 256) {
            int row = s >> 2, seg = s & 3;
            async_g2l_b128(&lsA[buf][row][seg * 8],
                           A + (size_t)(m0 + row) * K + k0 + seg * 8);
        }
        // B tile: 32 rows x 128 cols, scatter K-transposed (needs VGPR path)
#pragma unroll
        for (int s = tid; s < 512; s += 256) {
            int krow = s >> 4, cseg = s & 15;
            uint4 t = *(const uint4*)(Bm + (size_t)(k0 + krow) * Nn + n0 + cseg * 8);
            const __bf16* hs = reinterpret_cast<const __bf16*>(&t);
#pragma unroll
            for (int e = 0; e < 8; ++e) lsB[buf][cseg * 8 + e][krow] = hs[e];
        }
    };

    const int nk = K >> 5;
    load_tiles(0, 0);
    for (int i = 0; i < nk; ++i) {
        wait_async0();
        __syncthreads();
        if (i + 1 < nk) load_tiles((i + 1) << 5, (i + 1) & 1);

        const int buf = i & 1;
        bf16x16 aF[2], bF[4];
#pragma unroll
        for (int mi = 0; mi < 2; ++mi)
            aF[mi] = load_frag(&lsA[buf][0][0], 40, wm * 32 + mi * 16, 0);
#pragma unroll
        for (int ni = 0; ni < 4; ++ni)
            bF[ni] = load_frag(&lsB[buf][0][0], 40, wn * 64 + ni * 16, 0);
#pragma unroll
        for (int mi = 0; mi < 2; ++mi)
#pragma unroll
            for (int ni = 0; ni < 4; ++ni)
                acc[mi][ni] = __builtin_amdgcn_wmma_f32_16x16x32_bf16(
                    false, aF[mi], false, bF[ni], (short)0, acc[mi][ni], false, false);
    }

    // Epilogue. C/D layout: lane<16 -> M = r, lane>=16 -> M = 8+r ; N = lane&15
    const int lane = tid & 31;
    const int half = lane >> 4;
    const int lcol = lane & 15;
#pragma unroll
    for (int mi = 0; mi < 2; ++mi) {
#pragma unroll
        for (int ni = 0; ni < 4; ++ni) {
#pragma unroll
            for (int r = 0; r < 8; ++r) {
                int gr = m0 + wm * 32 + mi * 16 + half * 8 + r;
                int gc = n0 + wn * 64 + ni * 16 + lcol;
                float v = acc[mi][ni][r];
                if (BIAS)     v += bias[gc];
                if (GELU_ACT) v = gelu_tanh(v);
                if (RESID)    v += resid[(size_t)gr * Nn + gc];
                if (OUT_BF16) outB[(size_t)gr * Nn + gc] = (__bf16)v;
                else          outF[(size_t)gr * Nn + gc] = v;
            }
        }
    }
}

// ---------------------------------------------------------------------------
// Attention: one block per (b, h, 128-row Q tile). Streams 64-key tiles:
//   S = Q K^T * scale ; P = exp(S) (reference has no max-subtraction)
//   rowsum += sum_j P ; O += P V ; final O /= rowsum.
// Q/K tiles are async-DMA'd into LDS; V needs a transpose scatter.
// ---------------------------------------------------------------------------
__global__ __launch_bounds__(256) void attn_kernel(const __bf16* __restrict__ Q,
                                                   const __bf16* __restrict__ Kmat,
                                                   const __bf16* __restrict__ Vmat,
                                                   __bf16* __restrict__ O) {
    __shared__ __bf16 lsQ[128][72];   // [row][dh]
    __shared__ __bf16 lsK[64][72];    // [key][dh]  (B-transposed layout for QK^T)
    __shared__ __bf16 lsVt[64][72];   // [dh][key]  (B-transposed layout for PV)
    __shared__ __bf16 lsP[128][72];   // [row][key]
    __shared__ float  rowsumAcc[128];

    const int r0 = blockIdx.x * 128;
    const int h  = blockIdx.y;
    const int b  = blockIdx.z;
    const int tid  = threadIdx.x;
    const int wave = tid >> 5;
    const int wm   = wave & 3;     // 4 x 32 rows
    const int wn   = wave >> 2;    // 2 x 32 cols
    const int lane = tid & 31;
    const int half = lane >> 4;
    const int lcol = lane & 15;

    const __bf16* qbase = Q + ((size_t)(b * SEQ + r0)) * DMODEL + h * DHEAD;
#pragma unroll
    for (int s = tid; s < 1024; s += 256) {   // 128 rows x 8 uint4 segs
        int row = s >> 3, cs = s & 7;
        async_g2l_b128(&lsQ[row][cs * 8], qbase + (size_t)row * DMODEL + cs * 8);
    }
    if (tid < 128) rowsumAcc[tid] = 0.f;

    f32x8 oAcc[2][2];
#pragma unroll
    for (int mi = 0; mi < 2; ++mi)
#pragma unroll
        for (int ni = 0; ni < 2; ++ni) oAcc[mi][ni] = zero8();

    for (int j0 = 0; j0 < SEQ; j0 += 64) {
        __syncthreads();
        const __bf16* kbase = Kmat + ((size_t)(b * SEQ + j0)) * DMODEL + h * DHEAD;
        const __bf16* vbase = Vmat + ((size_t)(b * SEQ + j0)) * DMODEL + h * DHEAD;
#pragma unroll
        for (int s = tid; s < 512; s += 256) {  // K tile row-major, async DMA
            int kr = s >> 3, cs = s & 7;
            async_g2l_b128(&lsK[kr][cs * 8], kbase + (size_t)kr * DMODEL + cs * 8);
        }
#pragma unroll
        for (int s = tid; s < 512; s += 256) {  // V tile, scatter transposed
            int kr = s >> 3, cs = s & 7;
            uint4 t = *(const uint4*)(vbase + (size_t)kr * DMODEL + cs * 8);
            const __bf16* hs = reinterpret_cast<const __bf16*>(&t);
#pragma unroll
            for (int e = 0; e < 8; ++e) lsVt[cs * 8 + e][kr] = hs[e];
        }
        wait_async0();
        __syncthreads();

        // S = Q K^T  (M=128 rows, N=64 keys, K=64=dh)
        f32x8 sAcc[2][2];
#pragma unroll
        for (int mi = 0; mi < 2; ++mi)
#pragma unroll
            for (int ni = 0; ni < 2; ++ni) sAcc[mi][ni] = zero8();
#pragma unroll
        for (int kk = 0; kk < 64; kk += 32) {
            bf16x16 aF[2], bF[2];
#pragma unroll
            for (int mi = 0; mi < 2; ++mi)
                aF[mi] = load_frag(&lsQ[0][0], 72, wm * 32 + mi * 16, kk);
#pragma unroll
            for (int ni = 0; ni < 2; ++ni)
                bF[ni] = load_frag(&lsK[0][0], 72, wn * 32 + ni * 16, kk);
#pragma unroll
            for (int mi = 0; mi < 2; ++mi)
#pragma unroll
                for (int ni = 0; ni < 2; ++ni)
                    sAcc[mi][ni] = __builtin_amdgcn_wmma_f32_16x16x32_bf16(
                        false, aF[mi], false, bF[ni], (short)0, sAcc[mi][ni], false, false);
        }
        // P = exp(scale * S) into LDS
#pragma unroll
        for (int mi = 0; mi < 2; ++mi)
#pragma unroll
            for (int ni = 0; ni < 2; ++ni)
#pragma unroll
                for (int r = 0; r < 8; ++r) {
                    int row = wm * 32 + mi * 16 + half * 8 + r;
                    int col = wn * 32 + ni * 16 + lcol;
                    lsP[row][col] = (__bf16)expf(sAcc[mi][ni][r] * ATTN_SCALE);
                }
        __syncthreads();
        // rowsum accumulate (threads 0..127 own one row each)
        if (tid < 128) {
            float s = 0.f;
#pragma unroll
            for (int c = 0; c < 64; ++c) s += (float)lsP[tid][c];
            rowsumAcc[tid] += s;
        }
        // O += P V  (M=128 rows, N=64=dh, K=64 keys)
#pragma unroll
        for (int kk = 0; kk < 64; kk += 32) {
            bf16x16 aF[2], bF[2];
#pragma unroll
            for (int mi = 0; mi < 2; ++mi)
                aF[mi] = load_frag(&lsP[0][0], 72, wm * 32 + mi * 16, kk);
#pragma unroll
            for (int ni = 0; ni < 2; ++ni)
                bF[ni] = load_frag(&lsVt[0][0], 72, wn * 32 + ni * 16, kk);
#pragma unroll
            for (int mi = 0; mi < 2; ++mi)
#pragma unroll
                for (int ni = 0; ni < 2; ++ni)
                    oAcc[mi][ni] = __builtin_amdgcn_wmma_f32_16x16x32_bf16(
                        false, aF[mi], false, bF[ni], (short)0, oAcc[mi][ni], false, false);
        }
    }
    __syncthreads();

    __bf16* obase = O + ((size_t)(b * SEQ + r0)) * DMODEL + h * DHEAD;
#pragma unroll
    for (int mi = 0; mi < 2; ++mi)
#pragma unroll
        for (int ni = 0; ni < 2; ++ni)
#pragma unroll
            for (int r = 0; r < 8; ++r) {
                int row = wm * 32 + mi * 16 + half * 8 + r;
                int col = wn * 32 + ni * 16 + lcol;
                obase[(size_t)row * DMODEL + col] = (__bf16)(oAcc[mi][ni][r] / rowsumAcc[row]);
            }
}

// ---------------------------------------------------------------------------
// Mean pool over tokens + head LayerNorm + classifier (tiny: 8x512x1000, f32).
// One block per batch row, 512 threads (one per feature column).
// ---------------------------------------------------------------------------
__global__ __launch_bounds__(512) void pool_head_kernel(const float* __restrict__ x3,
                                                        const float* __restrict__ g,
                                                        const float* __restrict__ bta,
                                                        const float* __restrict__ Wh,
                                                        const float* __restrict__ bh,
                                                        float* __restrict__ out) {
    __shared__ float red[512];
    __shared__ float pool[512];
    __shared__ float stats[2];
    const int b = blockIdx.x;
    const int tid = threadIdx.x;

    float s = 0.f;
    const float* base = x3 + (size_t)b * SEQ * DMODEL + tid;
    for (int n = 0; n < SEQ; ++n) s += base[(size_t)n * DMODEL];
    float p = s * (1.f / SEQ);

    red[tid] = p;
    __syncthreads();
#pragma unroll
    for (int off = 256; off > 0; off >>= 1) {
        if (tid < off) red[tid] += red[tid + off];
        __syncthreads();
    }
    if (tid == 0) stats[0] = red[0] * (1.f / DMODEL);
    __syncthreads();
    const float mu = stats[0];
    float d = p - mu;
    red[tid] = d * d;
    __syncthreads();
#pragma unroll
    for (int off = 256; off > 0; off >>= 1) {
        if (tid < off) red[tid] += red[tid + off];
        __syncthreads();
    }
    if (tid == 0) stats[1] = rsqrtf(red[0] * (1.f / DMODEL) + 1e-5f);
    __syncthreads();
    pool[tid] = d * stats[1] * g[tid] + bta[tid];
    __syncthreads();

    for (int ci = tid; ci < NCLS; ci += 512) {
        float acc = bh[ci];
        for (int k = 0; k < DMODEL; ++k) acc += pool[k] * Wh[(size_t)k * NCLS + ci];
        out[(size_t)b * NCLS + ci] = acc;
    }
}

// ---------------------------------------------------------------------------
// Host-side orchestration
// ---------------------------------------------------------------------------
extern "C" void kernel_launch(void* const* d_in, const int* in_sizes, int n_in,
                              void* d_out, int out_size, void* d_ws, size_t ws_size,
                              hipStream_t stream) {
    (void)in_sizes; (void)n_in; (void)out_size; (void)ws_size;
    const float* x     = (const float*)d_in[0];
    const float* mask  = (const float*)d_in[1];
    const float* ln1_g = (const float*)d_in[2];
    const float* ln1_b = (const float*)d_in[3];
    const float* Wq    = (const float*)d_in[4];
    const float* bq    = (const float*)d_in[5];
    const float* Wk    = (const float*)d_in[6];
    const float* bk_p  = (const float*)d_in[7];
    const float* Wv    = (const float*)d_in[8];
    const float* bv    = (const float*)d_in[9];
    const float* Wo    = (const float*)d_in[10];
    const float* bo    = (const float*)d_in[11];
    const float* ln2_g = (const float*)d_in[12];
    const float* ln2_b = (const float*)d_in[13];
    const float* W1    = (const float*)d_in[14];
    const float* b1    = (const float*)d_in[15];
    const float* W2    = (const float*)d_in[16];
    const float* b2    = (const float*)d_in[17];
    const float* lnh_g = (const float*)d_in[18];
    const float* lnh_b = (const float*)d_in[19];
    const float* Wh    = (const float*)d_in[20];
    const float* bh    = (const float*)d_in[21];

    char* ws = (char*)d_ws;
    const size_t MB = 1ull << 20;
    // Workspace map (aliasing is ordered by the stream):
    //   [ 0,16MB)  resid (f32)  -- reused as x3 after the Wo projection
    //   [16,32MB)  x2 (f32)
    //   [32,40MB)  xn (bf16)    -- reused as xn2
    //   [40,72MB)  bufQ/bufK/bufV/bufAO (bf16, 8MB each) -- reused as h1 (32MB)
    //   [72MB,..)  bf16 weights
    float*  resid  = (float*) (ws + 0);
    float*  x2     = (float*) (ws + 16 * MB);
    __bf16* xn     = (__bf16*)(ws + 32 * MB);
    __bf16* bufQ   = (__bf16*)(ws + 40 * MB);
    __bf16* bufK   = (__bf16*)(ws + 48 * MB);
    __bf16* bufV   = (__bf16*)(ws + 56 * MB);
    __bf16* bufAO  = (__bf16*)(ws + 64 * MB);
    __bf16* h1     = bufQ;
    float*  x3     = resid;
    __bf16* wqB    = (__bf16*)(ws + 72 * MB);
    __bf16* wkB    = wqB + DMODEL * DMODEL;
    __bf16* wvB    = wkB + DMODEL * DMODEL;
    __bf16* woB    = wvB + DMODEL * DMODEL;
    __bf16* w1B    = woB + DMODEL * DMODEL;
    __bf16* w2B    = w1B + DMODEL * DFF;

    const int nDD = DMODEL * DMODEL;   // 262144
    const int nDF = DMODEL * DFF;      // 1048576
    cvt_bf16_kernel<<<(nDD + 255) / 256, 256, 0, stream>>>(Wq, wqB, nDD);
    cvt_bf16_kernel<<<(nDD + 255) / 256, 256, 0, stream>>>(Wk, wkB, nDD);
    cvt_bf16_kernel<<<(nDD + 255) / 256, 256, 0, stream>>>(Wv, wvB, nDD);
    cvt_bf16_kernel<<<(nDD + 255) / 256, 256, 0, stream>>>(Wo, woB, nDD);
    cvt_bf16_kernel<<<(nDF + 255) / 256, 256, 0, stream>>>(W1, w1B, nDF);
    cvt_bf16_kernel<<<(nDF + 255) / 256, 256, 0, stream>>>(W2, w2B, nDF);

    // mask * x -> residual ; LN1 -> xn (bf16)
    ln_kernel<true><<<MROWS, 256, 0, stream>>>(x, mask, resid, ln1_g, ln1_b, xn);

    // QKV projections
    dim3 gDD(DMODEL / 128, MROWS / 128);   // (4, 64)
    gemm_bf16_kernel<true, false, false, true><<<gDD, 256, 0, stream>>>(
        xn, wqB, bq, nullptr, nullptr, bufQ, MROWS, DMODEL, DMODEL);
    gemm_bf16_kernel<true, false, false, true><<<gDD, 256, 0, stream>>>(
        xn, wkB, bk_p, nullptr, nullptr, bufK, MROWS, DMODEL, DMODEL);
    gemm_bf16_kernel<true, false, false, true><<<gDD, 256, 0, stream>>>(
        xn, wvB, bv, nullptr, nullptr, bufV, MROWS, DMODEL, DMODEL);

    // Attention
    attn_kernel<<<dim3(SEQ / 128, HEADS, BATCH), 256, 0, stream>>>(bufQ, bufK, bufV, bufAO);

    // Output projection + residual -> x2 (f32)
    gemm_bf16_kernel<true, false, true, false><<<gDD, 256, 0, stream>>>(
        bufAO, woB, bo, resid, x2, nullptr, MROWS, DMODEL, DMODEL);

    // LN2 -> xn2 (bf16)
    ln_kernel<false><<<MROWS, 256, 0, stream>>>(x2, nullptr, nullptr, ln2_g, ln2_b, xn);

    // FFN1: GELU(xn2 @ W1 + b1) -> h1 (bf16)
    gemm_bf16_kernel<true, true, false, true><<<dim3(DFF / 128, MROWS / 128), 256, 0, stream>>>(
        xn, w1B, b1, nullptr, nullptr, h1, MROWS, DFF, DMODEL);

    // FFN2: h1 @ W2 + b2 + x2 -> x3 (f32)
    gemm_bf16_kernel<true, false, true, false><<<gDD, 256, 0, stream>>>(
        h1, w2B, b2, x2, x3, nullptr, MROWS, DMODEL, DFF);

    // Mean pool + head LN + classifier
    pool_head_kernel<<<BATCH, 512, 0, stream>>>(x3, lnh_g, lnh_b, Wh, bh, (float*)d_out);
}
